// Mamba3_25323127177601
// MI455X (gfx1250) — matrix operand order repaired and checked by reading, not compiled, not run on hardware
//
#include <hip/hip_runtime.h>
#include <hip/hip_bf16.h>
#include <math.h>

// ---- problem constants (from reference) ----
#define D_MODEL 1024
#define D_INNER 2048
#define NHEADS  32
#define HEADDIM 64
#define D_STATE 128
#define NROPE   32
#define LSEQ    2048
#define BATCH   2
#define NTOK    (BATCH*LSEQ)     // 4096
#define DP      4480             // D_IN_PROJ = 2*2048 + 2*128 + 3*32 + 32
#define OFF_Z    0
#define OFF_X    2048
#define OFF_B    4096
#define OFF_C    4224
#define OFF_DT   4352
#define OFF_A    4384
#define OFF_TR   4416
#define OFF_ROPE 4448
#define A_FLOOR 0.0001f
#define EPSV    1e-5f
#define CHUNK   16               // scan steps staged per TDM transfer

typedef __bf16 bf16;
typedef __attribute__((ext_vector_type(16))) __bf16 bf16x16;
typedef __attribute__((ext_vector_type(8)))  float  f32x8;
typedef __attribute__((ext_vector_type(4)))  unsigned int u32x4;
typedef __attribute__((ext_vector_type(8)))  int    i32x8;
typedef __attribute__((ext_vector_type(4)))  int    i32x4;

union FragBF { bf16x16 v; uint4 q[2]; };

__device__ __forceinline__ float softplus_f(float x) {
    return (x > 20.f) ? x : log1pf(__expf(x));
}

// LDS byte address of a __shared__ object: flat LDS aperture keeps the
// workgroup-relative byte offset in addr[31:0] (ISA 10.2 aperture mapping).
__device__ __forceinline__ unsigned lds_byte_addr(const void* p) {
    return (unsigned)(unsigned long long)p;
}

// ---------------------------------------------------------------------------
// Tensor Data Mover: async 2-D tile load Global -> LDS.
// D# built per ISA 08_async_tensor 8.3/8.4:
//   group0: [1:0]=count=1, [63:32]=lds_addr, [120:64]=global_addr, [127:126]=type=2
//   group1: [17:16]=data_size(2 => 4B), [79:48]=tensor_dim0, [111:80]=tensor_dim1,
//           [127:112]=tile_dim0, [143:128]=tile_dim1, [207:160]=tensor_dim0_stride
// Tensor dims are set equal to the tile window (no OOB path). Per-wave op.
// ---------------------------------------------------------------------------
__device__ __forceinline__ void tdm_load_2d(unsigned lds_byte, const void* gptr,
                                            unsigned width_elems, unsigned rows,
                                            unsigned long long row_stride_elems) {
    const unsigned long long ga = (unsigned long long)gptr;
    u32x4 g0;
    g0[0] = 1u;                                                    // count=1 user D#
    g0[1] = lds_byte;                                              // lds_addr
    g0[2] = (unsigned)(ga & 0xFFFFFFFFu);                          // global_addr lo
    g0[3] = (unsigned)((ga >> 32) & 0x01FFFFFFu) | 0x80000000u;    // ga[56:32] | type=2
    i32x8 g1;
    g1[0] = (int)(2u << 16);                                       // data_size = 4B
    g1[1] = (int)((width_elems & 0xFFFFu) << 16);                  // tensor_dim0[15:0]
    g1[2] = (int)((width_elems >> 16) | ((rows & 0xFFFFu) << 16)); // tdim0 hi | tdim1 lo
    g1[3] = (int)((rows >> 16) | ((width_elems & 0xFFFFu) << 16)); // tdim1 hi | tile_dim0
    g1[4] = (int)(rows & 0xFFFFu);                                 // tile_dim1 (tile_dim2=0)
    g1[5] = (int)(unsigned)(row_stride_elems & 0xFFFFFFFFull);     // dim0_stride lo32
    g1[6] = (int)((row_stride_elems >> 32) & 0xFFFFull);           // dim0_stride hi16
    g1[7] = 0;
    const i32x4 z4 = {0, 0, 0, 0};
#if __clang_major__ >= 23
    const i32x8 z8 = {0, 0, 0, 0, 0, 0, 0, 0};
    __builtin_amdgcn_tensor_load_to_lds(g0, g1, z4, z4, z8, 0);
#else
    __builtin_amdgcn_tensor_load_to_lds(g0, g1, z4, z4, 0);
#endif
}

// ---------------------------------------------------------------------------
// f32 -> bf16 elementwise convert
// ---------------------------------------------------------------------------
__global__ void k_f32_to_bf16(const float* __restrict__ src, bf16* __restrict__ dst, int n) {
    int i = blockIdx.x * blockDim.x + threadIdx.x;
    if (i < n) dst[i] = (bf16)src[i];
}

// ---------------------------------------------------------------------------
// C[M,N] = A[M,K] * B[N,K]^T   (A, B row-major bf16, K contiguous; C f32)
// block = 128 threads (4 waves). Wave w -> 16-row band; block tile 64M x 64N.
// Per wave per k-step: 1 A-frag (reused) x 4 B-frags -> 4 WMMA.
// ---------------------------------------------------------------------------
__global__ void __launch_bounds__(128)
k_gemm_nt(const bf16* __restrict__ A, const bf16* __restrict__ B,
          float* __restrict__ C, int M, int N, int K) {
    const int lane = threadIdx.x & 31;
    const int wave = threadIdx.x >> 5;
    const int m0 = blockIdx.y * 64 + wave * 16;
    const int n0 = blockIdx.x * 64;

    // A frag: lanes 0-15 hold M=lane, K 0..15; lanes 16-31 same M, K 16..31
    const int khalfA = (lane >> 4) * 16;
    // B frag: lanes 0-15 hold N=lane, K 0..15; lanes 16-31 N=lane-16, K 16..31
    const int koffB  = (lane >> 4) * 16;

    const bf16* arow  = A + (size_t)(m0 + (lane & 15)) * K;
    const bf16* brow0 = B + (size_t)(n0 + (lane & 15)) * K;
    const bf16* brow1 = brow0 + (size_t)16 * K;
    const bf16* brow2 = brow0 + (size_t)32 * K;
    const bf16* brow3 = brow0 + (size_t)48 * K;

    f32x8 acc0 = {}, acc1 = {}, acc2 = {}, acc3 = {};

    for (int k0 = 0; k0 < K; k0 += 32) {
        // speculative prefetch of the A row one unrolled block ahead
        __builtin_prefetch(arow + k0 + 256, 0, 0);

        FragBF a, b0, b1, b2, b3;
        a.q[0]  = *(const uint4*)(arow  + k0 + khalfA);
        a.q[1]  = *(const uint4*)(arow  + k0 + khalfA + 8);
        b0.q[0] = *(const uint4*)(brow0 + k0 + koffB);
        b0.q[1] = *(const uint4*)(brow0 + k0 + koffB + 8);
        b1.q[0] = *(const uint4*)(brow1 + k0 + koffB);
        b1.q[1] = *(const uint4*)(brow1 + k0 + koffB + 8);
        b2.q[0] = *(const uint4*)(brow2 + k0 + koffB);
        b2.q[1] = *(const uint4*)(brow2 + k0 + koffB + 8);
        b3.q[0] = *(const uint4*)(brow3 + k0 + koffB);
        b3.q[1] = *(const uint4*)(brow3 + k0 + koffB + 8);

        acc0 = __builtin_amdgcn_wmma_f32_16x16x32_bf16(false, a.v, false, b0.v, (short)0, acc0, false, false);
        acc1 = __builtin_amdgcn_wmma_f32_16x16x32_bf16(false, a.v, false, b1.v, (short)0, acc1, false, false);
        acc2 = __builtin_amdgcn_wmma_f32_16x16x32_bf16(false, a.v, false, b2.v, (short)0, acc2, false, false);
        acc3 = __builtin_amdgcn_wmma_f32_16x16x32_bf16(false, a.v, false, b3.v, (short)0, acc3, false, false);
    }

    // C/D layout: VGPR r -> row m0 + r + 8*(lane>=16); col = n0 + (lane&15) (+16 per acc)
    const int crow = m0 + ((lane >> 4) << 3);
    const int ccol = lane & 15;
    #pragma unroll
    for (int r = 0; r < 8; ++r) {
        float* cp = C + (size_t)(crow + r) * N + n0 + ccol;
        cp[0]  = acc0[r];
        cp[16] = acc1[r];
        cp[32] = acc2[r];
        cp[48] = acc3[r];
    }
}

// ---------------------------------------------------------------------------
// RoPE angle cumsum: grid = BATCH, block = NROPE lanes, sequential over L
// ---------------------------------------------------------------------------
__global__ void k_angles(const float* __restrict__ zx, float* __restrict__ ang) {
    const int b = blockIdx.x;
    const int j = threadIdx.x;                  // 0..31
    const float* p = zx + (size_t)b * LSEQ * DP + OFF_ROPE + j;
    float* o = ang + (size_t)b * LSEQ * NROPE + j;
    float acc = 0.f;
    for (int l = 0; l < LSEQ; ++l) {
        acc += p[(size_t)l * DP];
        o[(size_t)l * NROPE] = acc;
    }
}

// ---------------------------------------------------------------------------
// Per-token prep: RMSNorm(B/C) + RoPE(first 64) + scalar activations
// grid = NTOK, block = 128 (tid = state index d)
// ---------------------------------------------------------------------------
__global__ void __launch_bounds__(128)
k_prep(const float* __restrict__ zx, const float* __restrict__ ang,
       const float* __restrict__ dt_bias, const float* __restrict__ Bnw,
       const float* __restrict__ Cnw,
       float* __restrict__ Bn, float* __restrict__ Cn,
       float* __restrict__ dtv, float* __restrict__ adtv, float* __restrict__ trv) {
    const int bl  = blockIdx.x;
    const int tid = threadIdx.x;                // 0..127
    const float* row = zx + (size_t)bl * DP;

    float vB = row[OFF_B + tid];
    float vC = row[OFF_C + tid];

    // sum of squares across 128 threads (wave reduce + LDS combine)
    float sB = vB * vB, sC = vC * vC;
    #pragma unroll
    for (int off = 16; off; off >>= 1) {
        sB += __shfl_xor(sB, off, 32);
        sC += __shfl_xor(sC, off, 32);
    }
    __shared__ float wsB[4], wsC[4];
    if ((tid & 31) == 0) { wsB[tid >> 5] = sB; wsC[tid >> 5] = sC; }
    __syncthreads();
    const float totB = wsB[0] + wsB[1] + wsB[2] + wsB[3];
    const float totC = wsC[0] + wsC[1] + wsC[2] + wsC[3];
    const float rB = rsqrtf(totB * (1.0f / D_STATE) + EPSV);
    const float rC = rsqrtf(totC * (1.0f / D_STATE) + EPSV);

    float nB = vB * rB * Bnw[tid];
    float nC = vC * rC * Cnw[tid];

    if (tid < 64) {  // RoPE on first 64 dims; pair (2j, 2j+1) uses angle j
        const int j = tid >> 1;
        const float a = ang[(size_t)bl * NROPE + j];
        const float c = cosf(a), s = sinf(a);
        const float pB = __shfl_xor(nB, 1, 32);
        const float pC = __shfl_xor(nC, 1, 32);
        if ((tid & 1) == 0) { nB = nB * c - pB * s; nC = nC * c - pC * s; }
        else                { nB = pB * s + nB * c; nC = pC * s + nC * c; }
    }
    Bn[(size_t)bl * D_STATE + tid] = nB;
    Cn[(size_t)bl * D_STATE + tid] = nC;

    if (tid < NHEADS) {
        const float dts = softplus_f(row[OFF_DT + tid] + dt_bias[tid]);
        const float av  = softplus_f(row[OFF_A + tid]) + A_FLOOR;
        dtv [(size_t)bl * NHEADS + tid] = dts;
        adtv[(size_t)bl * NHEADS + tid] = -dts * av;
        trv [(size_t)bl * NHEADS + tid] = 1.f / (1.f + __expf(-row[OFF_TR + tid]));
    }
}

// ---------------------------------------------------------------------------
// Sequential selective-state scan. grid = (NHEADS, BATCH), block = 256.
// Thread owns p = tid>>2 (head dim), d = (tid&3)*32 .. +31 (state dim).
// State h[64][128] + Bx_prev entirely in registers (64 f32 / thread).
// Input staging: Tensor Data Mover 2-D tile DMAs (Bn/Cn/x/z) into
// double-buffered LDS; wave 0 issues the next chunk's TDM while all waves
// compute the current one; completion via s_wait_tensorcnt + s_barrier.
// 4-way D_STATE reduction via intra-wave shfl_xor (owner lanes adjacent).
// ---------------------------------------------------------------------------
__global__ void __launch_bounds__(256)
k_scan(const float* __restrict__ zx, const float* __restrict__ Bn,
       const float* __restrict__ Cn, const float* __restrict__ dtv,
       const float* __restrict__ adtv, const float* __restrict__ trv,
       const float* __restrict__ B_bias, const float* __restrict__ C_bias,
       const float* __restrict__ D_skip, bf16* __restrict__ yg) {
    const int h   = blockIdx.x;
    const int b   = blockIdx.y;
    const int tid = threadIdx.x;
    const int p     = tid >> 2;
    const int dbase = (tid & 3) * 32;
    const int base_bl = b * LSEQ;

    __shared__ float sBb[D_STATE], sCb[D_STATE];
    __shared__ float sB[2][CHUNK][D_STATE], sC[2][CHUNK][D_STATE];
    __shared__ float sx[2][CHUNK][HEADDIM], sz[2][CHUNK][HEADDIM];
    __shared__ float sdt[2][CHUNK], sadt[2][CHUNK], str[2][CHUNK];

    if (tid < D_STATE) {
        sBb[tid] = B_bias[h * D_STATE + tid];
        sCb[tid] = C_bias[h * D_STATE + tid];
    }
    const float dsk = D_skip[h];
    __syncthreads();

    // per-head bias slices kept in registers (removes 2 LDS reads per FMA)
    float bb[32], cb[32];
    #pragma unroll
    for (int j = 0; j < 32; ++j) { bb[j] = sBb[dbase + j]; cb[j] = sCb[dbase + j]; }

    float hst[32], bxp[32];
    #pragma unroll
    for (int j = 0; j < 32; ++j) { hst[j] = 0.f; bxp[j] = 0.f; }

    const int nchunks = LSEQ / CHUNK;

    // ---- prologue: stage chunk 0 into buffer 0 ----
    if (tid < CHUNK) {
        const size_t bl = (size_t)(base_bl + tid);
        sdt [0][tid] = dtv [bl * NHEADS + h];
        sadt[0][tid] = adtv[bl * NHEADS + h];
        str [0][tid] = trv [bl * NHEADS + h];
    }
    if (tid < 32) {   // wave 0 issues the TDM descriptors (per-wave op)
        const size_t bl0 = (size_t)base_bl;
        tdm_load_2d(lds_byte_addr(&sB[0][0][0]), Bn + bl0 * D_STATE, D_STATE, CHUNK, D_STATE);
        tdm_load_2d(lds_byte_addr(&sC[0][0][0]), Cn + bl0 * D_STATE, D_STATE, CHUNK, D_STATE);
        tdm_load_2d(lds_byte_addr(&sx[0][0][0]), zx + bl0 * DP + OFF_X + h * HEADDIM, HEADDIM, CHUNK, DP);
        tdm_load_2d(lds_byte_addr(&sz[0][0][0]), zx + bl0 * DP + OFF_Z + h * HEADDIM, HEADDIM, CHUNK, DP);
    }

    for (int ci = 0; ci < nchunks; ++ci) {
        const int buf = ci & 1;
        if (tid < 32) __builtin_amdgcn_s_wait_tensorcnt(0);   // chunk ci landed in LDS
        __syncthreads();   // data visible to all waves; previous chunk's compute done

        if (ci + 1 < nchunks) {  // prefetch chunk ci+1 into the other buffer
            const int nb = buf ^ 1;
            if (tid < CHUNK) {
                const size_t bl = (size_t)(base_bl + (ci + 1) * CHUNK + tid);
                sdt [nb][tid] = dtv [bl * NHEADS + h];
                sadt[nb][tid] = adtv[bl * NHEADS + h];
                str [nb][tid] = trv [bl * NHEADS + h];
            }
            if (tid < 32) {
                const size_t bl0 = (size_t)(base_bl + (ci + 1) * CHUNK);
                tdm_load_2d(lds_byte_addr(&sB[nb][0][0]), Bn + bl0 * D_STATE, D_STATE, CHUNK, D_STATE);
                tdm_load_2d(lds_byte_addr(&sC[nb][0][0]), Cn + bl0 * D_STATE, D_STATE, CHUNK, D_STATE);
                tdm_load_2d(lds_byte_addr(&sx[nb][0][0]), zx + bl0 * DP + OFF_X + h * HEADDIM, HEADDIM, CHUNK, DP);
                tdm_load_2d(lds_byte_addr(&sz[nb][0][0]), zx + bl0 * DP + OFF_Z + h * HEADDIM, HEADDIM, CHUNK, DP);
            }
        }

        for (int t = 0; t < CHUNK; ++t) {
            const float dt    = sdt[buf][t];
            const float decay = __expf(sadt[buf][t]);
            const float trh   = 0.5f * str[buf][t];   // blend = Bx + 0.5*tr*(Bxp - Bx)
            const float xp    = sx[buf][t][p];
            float psum = 0.f;
            #pragma unroll
            for (int j = 0; j < 32; ++j) {
                const float Bt    = sB[buf][t][dbase + j] * bb[j];
                const float Bx    = xp * Bt;
                const float blend = Bx + trh * (bxp[j] - Bx);
                const float hv    = decay * hst[j] + dt * blend;
                hst[j] = hv;
                bxp[j] = Bx;
                psum = fmaf(sC[buf][t][dbase + j] * cb[j], hv, psum);
            }
            psum += __shfl_xor(psum, 1, 32);
            psum += __shfl_xor(psum, 2, 32);
            if ((tid & 3) == 0) {
                const float y  = psum + dsk * xp;
                const float zp = sz[buf][t][p];
                const float g  = zp / (1.f + __expf(-zp));   // silu(z)
                yg[(size_t)(base_bl + ci * CHUNK + t) * D_INNER + h * HEADDIM + p] = (bf16)(y * g);
            }
        }
    }
}

// ---------------------------------------------------------------------------
extern "C" void kernel_launch(void* const* d_in, const int* in_sizes, int n_in,
                              void* d_out, int out_size, void* d_ws, size_t ws_size,
                              hipStream_t stream) {
    (void)in_sizes; (void)n_in; (void)out_size; (void)ws_size;
    const float* u       = (const float*)d_in[0];
    const float* W_in    = (const float*)d_in[1];
    const float* dt_bias = (const float*)d_in[2];
    const float* B_bias  = (const float*)d_in[3];
    const float* C_bias  = (const float*)d_in[4];
    const float* B_nw    = (const float*)d_in[5];
    const float* C_nw    = (const float*)d_in[6];
    const float* D_skip  = (const float*)d_in[7];
    const float* W_out   = (const float*)d_in[8];
    float* out = (float*)d_out;

    char* ws = (char*)d_ws;
    size_t off = 0;
    auto carve = [&](size_t bytes) {
        void* p = ws + off;
        off += (bytes + 255) & ~(size_t)255;
        return p;
    };
    float* zx   = (float*)carve((size_t)NTOK * DP * 4);           // 73.4 MB
    bf16*  u16  = (bf16* )carve((size_t)NTOK * D_MODEL * 2);      //  8.4 MB
    bf16*  Wi16 = (bf16* )carve((size_t)DP * D_MODEL * 2);        //  9.2 MB
    bf16*  Wo16 = (bf16* )carve((size_t)D_MODEL * D_INNER * 2);   //  4.2 MB
    float* angv = (float*)carve((size_t)NTOK * NROPE * 4);
    float* Bn   = (float*)carve((size_t)NTOK * D_STATE * 4);
    float* Cn   = (float*)carve((size_t)NTOK * D_STATE * 4);
    float* dtv  = (float*)carve((size_t)NTOK * NHEADS * 4);
    float* adtv = (float*)carve((size_t)NTOK * NHEADS * 4);
    float* trv  = (float*)carve((size_t)NTOK * NHEADS * 4);
    bf16*  yg   = (bf16* )carve((size_t)NTOK * D_INNER * 2);      // 16.8 MB

    const int n1 = NTOK * D_MODEL;
    const int n2 = DP * D_MODEL;
    const int n3 = D_MODEL * D_INNER;
    k_f32_to_bf16<<<(n1 + 255) / 256, 256, 0, stream>>>(u, u16, n1);
    k_f32_to_bf16<<<(n2 + 255) / 256, 256, 0, stream>>>(W_in, Wi16, n2);
    k_f32_to_bf16<<<(n3 + 255) / 256, 256, 0, stream>>>(W_out, Wo16, n3);

    // zxbcdt = u @ W_in^T  : M=4096, N=4480, K=1024
    k_gemm_nt<<<dim3(DP / 64, NTOK / 64), 128, 0, stream>>>(u16, Wi16, zx, NTOK, DP, D_MODEL);

    k_angles<<<BATCH, NROPE, 0, stream>>>(zx, angv);
    k_prep<<<NTOK, 128, 0, stream>>>(zx, angv, dt_bias, B_nw, C_nw, Bn, Cn, dtv, adtv, trv);
    k_scan<<<dim3(NHEADS, BATCH), 256, 0, stream>>>(zx, Bn, Cn, dtv, adtv, trv,
                                                    B_bias, C_bias, D_skip, yg);

    // out = yg @ W_out^T  : M=4096, N=1024, K=2048
    k_gemm_nt<<<dim3(D_MODEL / 64, NTOK / 64), 128, 0, stream>>>(yg, Wo16, out, NTOK, D_MODEL, D_INNER);
}
